// BasicDeformConv2d_88794153877750
// MI455X (gfx1250) — matrix operand order repaired and checked by reading, not compiled, not run on hardware
//
#include <hip/hip_runtime.h>
#include <hip/hip_bf16.h>

typedef _Float16 v16h __attribute__((ext_vector_type(16)));
typedef _Float16 v8h  __attribute__((ext_vector_type(8)));
typedef float    v8f  __attribute__((ext_vector_type(8)));
typedef float    v4f  __attribute__((ext_vector_type(4)));

// Problem constants
#define BATCH 16
#define CIN   128
#define HH    64
#define WW    64
#define COUT  256
#define KK9   9
#define KDIM  (CIN * KK9)       // 1152
#define NPIX  (BATCH * HH * WW) // 65536
#define ASTRIDE 40              // LDS A row stride in halves (16B aligned, padded)

// ---------------------------------------------------------------------------
// Kernel 0a: repack deformable weights f32 (o,c,kh,kw) -> f16 [o][kpos*128+c]
// ---------------------------------------------------------------------------
__global__ void repack_def_k(const float* __restrict__ w_def, _Float16* __restrict__ Wt) {
    int idx = blockIdx.x * 256 + threadIdx.x;           // COUT*KDIM = 294912
    if (idx >= COUT * KDIM) return;
    int o = idx / KDIM, kk = idx % KDIM;
    int kpos = kk >> 7, c = kk & 127;
    Wt[idx] = (_Float16)w_def[(o * CIN + c) * KK9 + kpos];
}

// ---------------------------------------------------------------------------
// Kernel 0b: repack offset-conv weights f32 (18,c,kh,kw) -> f16 [32][kpos*128+c]
// ---------------------------------------------------------------------------
__global__ void repack_off_k(const float* __restrict__ w_off, _Float16* __restrict__ Wt) {
    int idx = blockIdx.x * 256 + threadIdx.x;           // 32*KDIM = 36864
    if (idx >= 32 * KDIM) return;
    int o = idx / KDIM, kk = idx % KDIM;
    int kpos = kk >> 7, c = kk & 127;
    Wt[idx] = (o < 18) ? (_Float16)w_off[(o * CIN + c) * KK9 + kpos] : (_Float16)0.0f;
}

// ---------------------------------------------------------------------------
// Kernel 1: offset conv via implicit-GEMM WMMA.  M-tile = 64 px (one image
// row), N = 32 (18 valid).  Output layout off_ws[m][18] (pixel-major, f32).
// A-tile gather is BRANCH-FREE: clamped address + float mask, so the 8
// channel loads form one vmem clause with a single wait.
// ---------------------------------------------------------------------------
__global__ __launch_bounds__(256) void offconv_k(const float* __restrict__ x,
                                                 const float* __restrict__ b_off,
                                                 const _Float16* __restrict__ Wt,
                                                 float* __restrict__ off_out) {
    __shared__ __align__(16) _Float16 Alds[64][ASTRIDE];

    const int t       = threadIdx.x;
    const int m_base  = blockIdx.x * 64;
    const int b       = m_base >> 12;
    const int hw_base = m_base & 4095;
    const int h       = hw_base >> 6;
    const int px      = t & 63;           // pixel in tile == w coordinate
    const int g       = t >> 6;           // channel group (8 ch each)
    const int lane    = t & 31;
    const int wv      = t >> 5;
    const int rowtile = wv >> 1;          // 4 row tiles of 16 px
    const int coltile = wv & 1;           // 2 col tiles of 16 o
    const int r       = lane & 15;
    const int hf      = lane >> 4;

    const float* xb = x + (size_t)b * (CIN * HH * WW);

    v8f acc;
#pragma unroll
    for (int i = 0; i < 8; ++i) acc[i] = 0.0f;

    for (int kc = 0; kc < 36; ++kc) {
        const int kpos = kc >> 2;
        const int c0   = (kc & 3) << 5;
        const int kh   = kpos / 3, kw = kpos % 3;
        const int yy   = h - 1 + kh;
        const int xx   = px - 1 + kw;
        // branch-free: clamp address, scale by validity mask
        const float mask = ((yy >= 0) & (yy < HH) & (xx >= 0) & (xx < WW)) ? 1.0f : 0.0f;
        const int yyc = min(max(yy, 0), HH - 1);
        const int xxc = min(max(xx, 0), WW - 1);
        const float* src = xb + (size_t)(c0 + g * 8) * (HH * WW) + yyc * WW + xxc;
        v8h hv;
#pragma unroll
        for (int i = 0; i < 8; ++i) {
            float v = mask * src[i * (HH * WW)];
            hv[i] = (_Float16)v;
        }
        *(v8h*)&Alds[px][g * 8] = hv;
        __syncthreads();

        v16h a, bf;
        const _Float16* ap = &Alds[rowtile * 16 + r][hf * 8];
        ((v8h*)&a)[0] = *(const v8h*)ap;
        ((v8h*)&a)[1] = *(const v8h*)(ap + 16);
        const _Float16* bp = Wt + (size_t)(coltile * 16 + r) * KDIM + kc * 32 + hf * 8;
        ((v8h*)&bf)[0] = *(const v8h*)bp;
        ((v8h*)&bf)[1] = *(const v8h*)(bp + 16);
        acc = __builtin_amdgcn_wmma_f32_16x16x32_f16(false, a, false, bf,
                                                     (short)0, acc, false, false);
        __syncthreads();
    }

    const int o = coltile * 16 + r;
    if (o < 18) {
        const float bias = b_off[o];
        const int m0 = m_base + rowtile * 16 + hf * 8;
#pragma unroll
        for (int i = 0; i < 8; ++i)
            off_out[(size_t)(m0 + i) * 18 + o] = acc[i] + bias;
    }
}

// ---------------------------------------------------------------------------
// Kernel 2: fused bilinear-sample + implicit-GEMM WMMA main conv.
// Block tile: 64 px (one image row) x 256 Cout.  8 waves: wave = (rowtile 0-3,
// colgroup 0-1), each wave = 16 px x 128 o = 8 wmma tiles.
// ---------------------------------------------------------------------------
__global__ __launch_bounds__(256) void dconv_main_k(const float* __restrict__ x,
                                                    const float* __restrict__ off_in,
                                                    const _Float16* __restrict__ Wt,
                                                    float* __restrict__ out) {
    __shared__ __align__(16) _Float16 Alds[64][ASTRIDE];
    __shared__ float offs[64][18];

    const int t       = threadIdx.x;
    const int m_base  = blockIdx.x * 64;
    const int b       = m_base >> 12;
    const int hw_base = m_base & 4095;
    const int h       = hw_base >> 6;
    const int px      = t & 63;           // pixel in tile == w coordinate
    const int g       = t >> 6;           // channel group of 8
    const int lane    = t & 31;
    const int wv      = t >> 5;
    const int rowtile = wv >> 1;          // 0..3
    const int colg    = wv & 1;           // 0..1 -> o base 0 / 128
    const int r       = lane & 15;
    const int hf      = lane >> 4;

    const float* xb = x + (size_t)b * (CIN * HH * WW);

    // stage this tile's 18 offset channels per pixel into LDS
    for (int i = t; i < 64 * 18; i += 256)
        ((float*)offs)[i] = off_in[(size_t)m_base * 18 + i];

    v8f acc[8];
#pragma unroll
    for (int nb = 0; nb < 8; ++nb)
#pragma unroll
        for (int i = 0; i < 8; ++i) acc[nb][i] = 0.0f;

    __syncthreads();

    for (int kc = 0; kc < 36; ++kc) {
        const int kpos = kc >> 2;
        const int c0   = (kc & 3) << 5;
        const int kh   = kpos / 3, kw = kpos % 3;

        // bilinear setup (one tap, shared by 32 channels of this chunk)
        const float dy  = offs[px][kpos * 2 + 0];
        const float dx  = offs[px][kpos * 2 + 1];
        const float ysf = (float)(h - 1 + kh) + dy;
        const float xsf = (float)(px - 1 + kw) + dx;
        const float y0f = floorf(ysf), x0f = floorf(xsf);
        const float wy1 = ysf - y0f,  wx1 = xsf - x0f;
        const float wy0 = 1.0f - wy1, wx0 = 1.0f - wx1;
        const int y0 = (int)y0f, x0 = (int)x0f;
        const int y1 = y0 + 1,   x1 = x0 + 1;
        const float vy0 = (y0 >= 0 && y0 < HH) ? 1.0f : 0.0f;
        const float vy1 = (y1 >= 0 && y1 < HH) ? 1.0f : 0.0f;
        const float vx0 = (x0 >= 0 && x0 < WW) ? 1.0f : 0.0f;
        const float vx1 = (x1 >= 0 && x1 < WW) ? 1.0f : 0.0f;
        const float w00 = wy0 * wx0 * vy0 * vx0;
        const float w01 = wy0 * wx1 * vy0 * vx1;
        const float w10 = wy1 * wx0 * vy1 * vx0;
        const float w11 = wy1 * wx1 * vy1 * vx1;
        const int y0c = min(max(y0, 0), HH - 1), y1c = min(max(y1, 0), HH - 1);
        const int x0c = min(max(x0, 0), WW - 1), x1c = min(max(x1, 0), WW - 1);
        const int i00 = y0c * WW + x0c, i01 = y0c * WW + x1c;
        const int i10 = y1c * WW + x0c, i11 = y1c * WW + x1c;

        const float* s = xb + (size_t)(c0 + g * 8) * (HH * WW);
        v8h hv;
#pragma unroll
        for (int i = 0; i < 8; ++i) {
            const float* p = s + i * (HH * WW);
            float v = w00 * p[i00] + w01 * p[i01] + w10 * p[i10] + w11 * p[i11];
            hv[i] = (_Float16)v;
        }
        *(v8h*)&Alds[px][g * 8] = hv;
        __syncthreads();

        // A fragment (16-bit A 16x32 layout: lane<16 -> K 0-7,16-23; lane>=16 -> K 8-15,24-31)
        v16h a;
        const _Float16* ap = &Alds[rowtile * 16 + r][hf * 8];
        ((v8h*)&a)[0] = *(const v8h*)ap;
        ((v8h*)&a)[1] = *(const v8h*)(ap + 16);

        const int k0 = kc * 32 + hf * 8;
        // one advisory prefetch per chunk (next chunk's weight line)
        __builtin_prefetch(Wt + (size_t)(colg * 128 + r) * KDIM + k0 + 32, 0, 1);
#pragma unroll
        for (int nb = 0; nb < 8; ++nb) {
            const int o = colg * 128 + nb * 16 + r;
            const _Float16* bp = Wt + (size_t)o * KDIM + k0;
            v16h bf;
            ((v8h*)&bf)[0] = *(const v8h*)bp;
            ((v8h*)&bf)[1] = *(const v8h*)(bp + 16);
            acc[nb] = __builtin_amdgcn_wmma_f32_16x16x32_f16(false, a, false, bf,
                                                             (short)0, acc[nb], false, false);
        }
        __syncthreads();
    }

    // store: lane owns column o, 8 consecutive hw addresses -> 2x b128 per tile
    const int hw0 = hw_base + rowtile * 16 + hf * 8;
#pragma unroll
    for (int nb = 0; nb < 8; ++nb) {
        const int o = colg * 128 + nb * 16 + r;
        float* op = out + ((size_t)(b * COUT + o) << 12) + hw0;
        v4f lo = {acc[nb][0], acc[nb][1], acc[nb][2], acc[nb][3]};
        v4f hi = {acc[nb][4], acc[nb][5], acc[nb][6], acc[nb][7]};
        *(v4f*)op = lo;
        *(v4f*)(op + 4) = hi;
    }
}

// ---------------------------------------------------------------------------
// Launch
// ---------------------------------------------------------------------------
extern "C" void kernel_launch(void* const* d_in, const int* in_sizes, int n_in,
                              void* d_out, int out_size, void* d_ws, size_t ws_size,
                              hipStream_t stream) {
    (void)in_sizes; (void)n_in; (void)out_size; (void)ws_size;
    const float* x     = (const float*)d_in[0];
    const float* w_off = (const float*)d_in[1];
    const float* b_off = (const float*)d_in[2];
    const float* w_def = (const float*)d_in[3];
    float* out = (float*)d_out;

    // workspace layout (all 256B aligned):
    //   [0, 4718592)          off_ws : f32 [65536][18]
    //   [4718592, 5308416)    WtD    : f16 [256][1152]
    //   [5308416, 5382144)    WtO    : f16 [32][1152]
    char* ws = (char*)d_ws;
    float*     off_ws = (float*)ws;
    _Float16*  WtD    = (_Float16*)(ws + 4718592);
    _Float16*  WtO    = (_Float16*)(ws + 5308416);

    repack_def_k<<<(COUT * KDIM + 255) / 256, 256, 0, stream>>>(w_def, WtD);
    repack_off_k<<<(32 * KDIM + 255) / 256, 256, 0, stream>>>(w_off, WtO);
    offconv_k<<<NPIX / 64, 256, 0, stream>>>(x, b_off, WtO, off_ws);
    dconv_main_k<<<NPIX / 64, 256, 0, stream>>>(x, off_ws, WtD, out);
}
